// Attention_89902255440868
// MI455X (gfx1250) — matrix operand order repaired and checked by reading, not compiled, not run on hardware
//
#include <hip/hip_runtime.h>

typedef __attribute__((ext_vector_type(16))) _Float16 v16h;
typedef __attribute__((ext_vector_type(8)))  _Float16 v8h;
typedef __attribute__((ext_vector_type(4)))  _Float16 v4h;
typedef __attribute__((ext_vector_type(8)))  float    v8f;

#define HEADS  8
#define DHEAD  64
#define DIMX   512
#define DINNER 512
#define BB     4
#define NN     1024

// ---------------------------------------------------------------------------
// WMMA fragment loaders (wave32, v_wmma_f32_16x16x32_f16 layouts per ISA 7.12.2)
// A: 16x32 f16. lane L: row m=L&15; holds K = {kb..kb+7} and {kb+16..kb+23}, kb=(L>=16)*8
// B: 32x16 f16 from a row-major [n][k] tile. lane L: col n=L&15; K = {kb..kb+7} and {kb+16..kb+23}, kb=(L>=16)*8
// ld (in halves) must be a multiple of 8 so every v8h access is 16B aligned.
// ---------------------------------------------------------------------------
__device__ __forceinline__ v16h lds_load_A(const _Float16* t, int ld) {
    int lane = threadIdx.x & 31;
    int m  = lane & 15;
    int kb = (lane >> 4) << 3;
    v8h lo = *(const v8h*)(t + m * ld + kb);
    v8h hi = *(const v8h*)(t + m * ld + kb + 16);
    v16h r;
#pragma unroll
    for (int e = 0; e < 8; ++e) { r[e] = lo[e]; r[8 + e] = hi[e]; }
    return r;
}

__device__ __forceinline__ v16h lds_load_B(const _Float16* t, int ld) {
    int lane = threadIdx.x & 31;
    int n  = lane & 15;
    int kb = (lane >> 4) << 3;
    v8h lo = *(const v8h*)(t + n * ld + kb);
    v8h hi = *(const v8h*)(t + n * ld + kb + 16);
    v16h r;
#pragma unroll
    for (int e = 0; e < 8; ++e) { r[e] = lo[e]; r[8 + e] = hi[e]; }
    return r;
}

// ---------------------------------------------------------------------------
// DPP butterfly reduction across each 16-lane half (pure VALU, no LDS).
// quad_perm xor1 = 0xB1, quad_perm xor2 = 0x4E,
// ROW_HALF_MIRROR = 0x141 (== xor4 once quads uniform),
// ROW_MIRROR      = 0x140 (== xor8 once 8-groups uniform).
// ---------------------------------------------------------------------------
#define DPP_BFLY(v, ctrl)                                                   \
    __int_as_float(__builtin_amdgcn_update_dpp(                             \
        __float_as_int(v), __float_as_int(v), (ctrl), 0xf, 0xf, true))

__device__ __forceinline__ float red_max16(float v) {
    v = fmaxf(v, DPP_BFLY(v, 0xB1));
    v = fmaxf(v, DPP_BFLY(v, 0x4E));
    v = fmaxf(v, DPP_BFLY(v, 0x141));
    v = fmaxf(v, DPP_BFLY(v, 0x140));
    return v;
}
__device__ __forceinline__ float red_sum16(float v) {
    v += DPP_BFLY(v, 0xB1);
    v += DPP_BFLY(v, 0x4E);
    v += DPP_BFLY(v, 0x141);
    v += DPP_BFLY(v, 0x140);
    return v;
}

// ---------------------------------------------------------------------------
// f32 -> f16 conversion (4 elements/thread)
// ---------------------------------------------------------------------------
__global__ void cvt_f32_f16(const float4* __restrict__ src, v4h* __restrict__ dst, int n4) {
    int i = blockIdx.x * blockDim.x + threadIdx.x;
    if (i < n4) {
        float4 f = src[i];
        v4h h;
        h[0] = (_Float16)f.x; h[1] = (_Float16)f.y;
        h[2] = (_Float16)f.z; h[3] = (_Float16)f.w;
        dst[i] = h;
    }
}

// ---------------------------------------------------------------------------
// Tiled WMMA GEMM:  out[M,N] = epilogue( X[M,K] @ W[N,K]^T )
// 128 threads (4 waves), 128x64 output tile, K-step 32.
// Each wave: 2 A-fragments (rows wave*16 and 64+wave*16) x 4 shared B-fragments
// = 8 WMMAs per K-step.
// MODE 0: outH = f16( acc * scale )                      (q / kv projections)
// MODE 2: outF = sigmoid( acc + bias[n] )                (gate projection)
// MODE 3: A staged as attn*gate; outF = acc + bias[n]    (output projection)
// ---------------------------------------------------------------------------
template <int MODE>
__global__ __launch_bounds__(128)
void gemm_xwt(const _Float16* __restrict__ X, const float* __restrict__ gate,
              const _Float16* __restrict__ W, const float* __restrict__ bias,
              _Float16* __restrict__ outH, float* __restrict__ outF,
              int M, int N, int K, float scale) {
    const int LD = 40;                        // 32 + 8 halves pad (80B rows, 16B aligned)
    __shared__ _Float16 Xs[128 * LD];
    __shared__ _Float16 Ws[64 * LD];

    int n0   = blockIdx.x * 64;
    int m0   = blockIdx.y * 128;
    int tid  = threadIdx.x;
    int wave = tid >> 5;
    int lane = tid & 31;

    v8f acc[2][4];
#pragma unroll
    for (int mb = 0; mb < 2; ++mb)
#pragma unroll
        for (int i = 0; i < 4; ++i)
#pragma unroll
            for (int r = 0; r < 8; ++r) acc[mb][i][r] = 0.f;

    for (int k0 = 0; k0 < K; k0 += 32) {
        // ---- stage X tile: one full 32-half row per thread ----
        const _Float16* xsrc = X + (size_t)(m0 + tid) * K + k0;
        if (MODE == 3) {
            const float* gsrc = gate + (size_t)(m0 + tid) * K + k0;
#pragma unroll
            for (int s = 0; s < 4; ++s) {
                v8h xa = *(const v8h*)(xsrc + s * 8);
                v8h ga;
#pragma unroll
                for (int i = 0; i < 8; ++i)
                    ga[i] = (_Float16)((float)xa[i] * gsrc[s * 8 + i]);
                *(v8h*)(Xs + tid * LD + s * 8) = ga;
            }
        } else {
#pragma unroll
            for (int s = 0; s < 4; ++s)
                *(v8h*)(Xs + tid * LD + s * 8) = *(const v8h*)(xsrc + s * 8);
        }
        // ---- stage W tile: 64 rows x 32 halves ----
        {
            int row = tid >> 1;
            int seg = (tid & 1) * 16;
            const _Float16* wsrc = W + (size_t)(n0 + row) * K + k0 + seg;
            *(v8h*)(Ws + row * LD + seg)     = *(const v8h*)(wsrc);
            *(v8h*)(Ws + row * LD + seg + 8) = *(const v8h*)(wsrc + 8);
        }
        __syncthreads();

        v16h a0 = lds_load_A(Xs + (wave * 16) * LD, LD);
        v16h a1 = lds_load_A(Xs + (64 + wave * 16) * LD, LD);
#pragma unroll
        for (int nb = 0; nb < 4; ++nb) {
            v16h b = lds_load_B(Ws + nb * 16 * LD, LD);
            acc[0][nb] = __builtin_amdgcn_wmma_f32_16x16x32_f16(
                false, a0, false, b, (short)0, acc[0][nb], false, false);
            acc[1][nb] = __builtin_amdgcn_wmma_f32_16x16x32_f16(
                false, a1, false, b, (short)0, acc[1][nb], false, false);
        }
        __syncthreads();
    }

    // ---- epilogue (C layout: c[r] -> row 8*(lane>=16)+r, col lane&15) ----
#pragma unroll
    for (int mb = 0; mb < 2; ++mb) {
        int mrow = m0 + mb * 64 + wave * 16 + (lane >> 4) * 8;
        int nc0  = n0 + (lane & 15);
#pragma unroll
        for (int nb = 0; nb < 4; ++nb) {
            int nc = nc0 + nb * 16;
#pragma unroll
            for (int r = 0; r < 8; ++r) {
                float v = acc[mb][nb][r];
                size_t idx = (size_t)(mrow + r) * N + nc;
                if (MODE == 0)      outH[idx] = (_Float16)(v * scale);
                else if (MODE == 2) outF[idx] = 1.f / (1.f + __expf(-(v + bias[nc])));
                else                outF[idx] = v + bias[nc];
            }
        }
    }
}

// ---------------------------------------------------------------------------
// Flash attention: block = (64 query rows, head h, batch b); 4 waves; KV tiles
// of 32 keys streamed through LDS; online softmax via DPP; bias from HBM.
// ---------------------------------------------------------------------------
__global__ __launch_bounds__(128)
void flash_attn(const _Float16* __restrict__ q16, const _Float16* __restrict__ kv16,
                const float* __restrict__ bias, const unsigned char* __restrict__ mask,
                _Float16* __restrict__ outA) {
    const int LQ = 72, LK = 72, LV = 40, LP = 40;
    __shared__ _Float16 Qs[64 * LQ];
    __shared__ _Float16 Ks[32 * LK];
    __shared__ _Float16 Vts[64 * LV];      // transposed: Vts[d][j]
    __shared__ _Float16 Ps[4 * 16 * LP];   // per-wave P tiles

    int i0 = blockIdx.x * 64;
    int h  = blockIdx.y;
    int b  = blockIdx.z;
    int tid = threadIdx.x, lane = tid & 31, wave = tid >> 5;

    // ---- stage Q tile [64 x 64] once ----
    {
        int row = tid >> 1;
        int off = (tid & 1) * 32;
        const _Float16* src = q16 + (size_t)(b * NN + i0 + row) * DINNER + h * DHEAD + off;
#pragma unroll
        for (int s = 0; s < 4; ++s)
            *(v8h*)(Qs + row * LQ + off + s * 8) = *(const v8h*)(src + s * 8);
    }
    __syncthreads();
    v16h qf0 = lds_load_A(Qs + wave * 16 * LQ, LQ);        // d = 0..31
    v16h qf1 = lds_load_A(Qs + wave * 16 * LQ + 32, LQ);   // d = 32..63

    float mrow[8], lrow[8];
#pragma unroll
    for (int r = 0; r < 8; ++r) { mrow[r] = -3.0e38f; lrow[r] = 0.f; }
    v8f o[4];
#pragma unroll
    for (int d = 0; d < 4; ++d)
#pragma unroll
        for (int r = 0; r < 8; ++r) o[d][r] = 0.f;

    for (int j0 = 0; j0 < NN; j0 += 32) {
        __syncthreads();   // previous iteration done with Ks/Vts
        // ---- stage K [32 x 64] row-major ----
        {
            int row = tid >> 2;
            int seg = (tid & 3) * 16;
            const _Float16* ksrc = kv16 + (size_t)(b * NN + j0 + row) * 1024 + h * DHEAD + seg;
            *(v8h*)(Ks + row * LK + seg)     = *(const v8h*)(ksrc);
            *(v8h*)(Ks + row * LK + seg + 8) = *(const v8h*)(ksrc + 8);
        }
        // ---- stage V transposed via 4x4 micro-tiles (all-b64 traffic) ----
        {
            int dg = (tid & 15) * 4;       // d group: 16 x 4
            int jg = (tid >> 4) * 4;       // j group:  8 x 4
            const _Float16* vbase =
                kv16 + (size_t)(b * NN + j0 + jg) * 1024 + 512 + h * DHEAD + dg;
            v4h r0 = *(const v4h*)(vbase);
            v4h r1 = *(const v4h*)(vbase + 1024);
            v4h r2 = *(const v4h*)(vbase + 2048);
            v4h r3 = *(const v4h*)(vbase + 3072);
#pragma unroll
            for (int dd = 0; dd < 4; ++dd) {
                v4h col;
                col[0] = r0[dd]; col[1] = r1[dd]; col[2] = r2[dd]; col[3] = r3[dd];
                *(v4h*)(Vts + (dg + dd) * LV + jg) = col;
            }
        }
        __syncthreads();

        // ---- S = Q K^T : 16 x 32 per wave (two 16x16 accumulators) ----
        v8f s0, s1;
#pragma unroll
        for (int r = 0; r < 8; ++r) { s0[r] = 0.f; s1[r] = 0.f; }
        {
            v16h bk = lds_load_B(Ks, LK);
            s0 = __builtin_amdgcn_wmma_f32_16x16x32_f16(false, qf0, false, bk, (short)0, s0, false, false);
            bk = lds_load_B(Ks + 32, LK);
            s0 = __builtin_amdgcn_wmma_f32_16x16x32_f16(false, qf1, false, bk, (short)0, s0, false, false);
            bk = lds_load_B(Ks + 16 * LK, LK);
            s1 = __builtin_amdgcn_wmma_f32_16x16x32_f16(false, qf0, false, bk, (short)0, s1, false, false);
            bk = lds_load_B(Ks + 16 * LK + 32, LK);
            s1 = __builtin_amdgcn_wmma_f32_16x16x32_f16(false, qf1, false, bk, (short)0, s1, false, false);
        }

        // ---- + attn_bias, key mask ----
        int ibase = i0 + wave * 16 + (lane >> 4) * 8;
        int jc    = j0 + (lane & 15);
#pragma unroll
        for (int r = 0; r < 8; ++r) {
            s0[r] += bias[((size_t)(h * NN + ibase + r)) * NN + jc];
            s1[r] += bias[((size_t)(h * NN + ibase + r)) * NN + jc + 16];
        }
        if (!mask[b * NN + jc]) {
#pragma unroll
            for (int r = 0; r < 8; ++r) s0[r] = -3.0e38f;
        }
        if (!mask[b * NN + jc + 16]) {
#pragma unroll
            for (int r = 0; r < 8; ++r) s1[r] = -3.0e38f;
        }

        // ---- online softmax (DPP butterfly across each 16-lane half) ----
        float corr[8];
#pragma unroll
        for (int r = 0; r < 8; ++r) {
            float pm = red_max16(fmaxf(s0[r], s1[r]));
            float mn = fmaxf(mrow[r], pm);
            corr[r]  = __expf(mrow[r] - mn);
            mrow[r]  = mn;
            s0[r] = __expf(s0[r] - mn);
            s1[r] = __expf(s1[r] - mn);
            lrow[r] = lrow[r] * corr[r] + red_sum16(s0[r] + s1[r]);
        }
#pragma unroll
        for (int d = 0; d < 4; ++d)
#pragma unroll
            for (int r = 0; r < 8; ++r) o[d][r] *= corr[r];

        // ---- P (f32 C-layout) -> f16 A-layout via per-wave LDS buffer ----
        _Float16* Pw = Ps + wave * 16 * LP;
#pragma unroll
        for (int r = 0; r < 8; ++r) {
            Pw[((lane >> 4) * 8 + r) * LP + (lane & 15)]      = (_Float16)s0[r];
            Pw[((lane >> 4) * 8 + r) * LP + 16 + (lane & 15)] = (_Float16)s1[r];
        }
        asm volatile("s_wait_dscnt 0" ::: "memory");   // wave-local LDS RAW
        v16h pf = lds_load_A(Pw, LP);
#pragma unroll
        for (int d = 0; d < 4; ++d) {
            v16h bv = lds_load_B(Vts + d * 16 * LV, LV);
            o[d] = __builtin_amdgcn_wmma_f32_16x16x32_f16(
                false, pf, false, bv, (short)0, o[d], false, false);
        }
    }

    // ---- normalize and write attn output (f16, [token][h*64+d]) ----
#pragma unroll
    for (int r = 0; r < 8; ++r) lrow[r] = (lrow[r] > 0.f) ? 1.f / lrow[r] : 0.f;
    int row0 = i0 + wave * 16 + (lane >> 4) * 8;
#pragma unroll
    for (int d = 0; d < 4; ++d)
#pragma unroll
        for (int r = 0; r < 8; ++r)
            outA[(size_t)(b * NN + row0 + r) * DINNER + h * DHEAD + d * 16 + (lane & 15)] =
                (_Float16)(o[d][r] * lrow[r]);
}

// ---------------------------------------------------------------------------
extern "C" void kernel_launch(void* const* d_in, const int* in_sizes, int n_in,
                              void* d_out, int out_size, void* d_ws, size_t ws_size,
                              hipStream_t stream) {
    const float*         x    = (const float*)d_in[0];
    const unsigned char* mask = (const unsigned char*)d_in[1];
    const float*         abia = (const float*)d_in[2];
    const float*         wq   = (const float*)d_in[3];
    const float*         wkv  = (const float*)d_in[4];
    const float*         wo   = (const float*)d_in[5];
    const float*         bo   = (const float*)d_in[6];
    const float*         wg   = (const float*)d_in[7];
    const float*         bg   = (const float*)d_in[8];
    float*               out  = (float*)d_out;

    const int M = BB * NN;                // 4096 tokens
    char* p = (char*)d_ws;
    _Float16* x16   = (_Float16*)p; p += (size_t)M * DIMX * 2;
    _Float16* wq16  = (_Float16*)p; p += (size_t)DINNER * DIMX * 2;
    _Float16* wkv16 = (_Float16*)p; p += (size_t)2 * DINNER * DIMX * 2;
    _Float16* wg16  = (_Float16*)p; p += (size_t)DINNER * DIMX * 2;
    _Float16* wo16  = (_Float16*)p; p += (size_t)DIMX * DINNER * 2;
    _Float16* q16   = (_Float16*)p; p += (size_t)M * DINNER * 2;
    _Float16* kv16  = (_Float16*)p; p += (size_t)M * 2 * DINNER * 2;
    _Float16* at16  = (_Float16*)p; p += (size_t)M * DINNER * 2;
    float*    g32   = (float*)p;    p += (size_t)M * DINNER * 4;

    auto cvt = [&](const float* s, _Float16* d, int n) {
        int n4 = n / 4;
        cvt_f32_f16<<<dim3((n4 + 255) / 256), dim3(256), 0, stream>>>(
            (const float4*)s, (v4h*)d, n4);
    };
    cvt(x,   x16,   M * DIMX);
    cvt(wq,  wq16,  DINNER * DIMX);
    cvt(wkv, wkv16, 2 * DINNER * DIMX);
    cvt(wg,  wg16,  DINNER * DIMX);
    cvt(wo,  wo16,  DIMX * DINNER);

    const float qscale = 0.125f;   // DIM_HEAD^-0.5
    // q = (x @ wq^T) * scale   -> f16
    gemm_xwt<0><<<dim3(DINNER / 64, M / 128), dim3(128), 0, stream>>>(
        x16, nullptr, wq16, nullptr, q16, nullptr, M, DINNER, DIMX, qscale);
    // kv = x @ wkv^T           -> f16 (k = cols 0..511, v = cols 512..1023)
    gemm_xwt<0><<<dim3(2 * DINNER / 64, M / 128), dim3(128), 0, stream>>>(
        x16, nullptr, wkv16, nullptr, kv16, nullptr, M, 2 * DINNER, DIMX, 1.0f);
    // g = sigmoid(x @ wg^T + bg) -> f32
    gemm_xwt<2><<<dim3(DINNER / 64, M / 128), dim3(128), 0, stream>>>(
        x16, nullptr, wg16, bg, nullptr, g32, M, DINNER, DIMX, 1.0f);
    // flash attention -> at16
    flash_attn<<<dim3(NN / 64, HEADS, BB), dim3(128), 0, stream>>>(
        q16, kv16, abia, mask, at16);
    // out = (at * g) @ wo^T + bo -> f32
    gemm_xwt<3><<<dim3(DIMX / 64, M / 128), dim3(128), 0, stream>>>(
        at16, g32, wo16, bo, nullptr, out, M, DIMX, DINNER, 1.0f);
}
